// GraphAttentionLayer_32796370272775
// MI455X (gfx1250) — matrix operand order repaired and checked by reading, not compiled, not run on hardware
//
#include <hip/hip_runtime.h>

// ---------- types ----------
typedef __attribute__((ext_vector_type(16))) __bf16        v16bf;
typedef __attribute__((ext_vector_type(8)))  float         v8f;
typedef __attribute__((ext_vector_type(4)))  unsigned int  u32x4;
typedef __attribute__((ext_vector_type(2)))  unsigned int  u32x2;

union ABFrag { v16bf v; u32x4 q[2]; };
union Pack8  { unsigned short s[8]; u32x4 q; };

#define NN      8192
#define DD      512
#define GALPHA  0.2f
#define NEGFILL -9.0e15f

__device__ __forceinline__ unsigned short f2bf(float f) {
    unsigned int u = __float_as_uint(f);
    u += 0x7FFFu + ((u >> 16) & 1u);           // round-to-nearest-even
    return (unsigned short)(u >> 16);
}

// ---------- 1) convert X (f32 -> bf16, same layout) ----------
__global__ void k_cvt_x(const float* __restrict__ x, unsigned short* __restrict__ xb, int n) {
    int i = blockIdx.x * 256 + threadIdx.x;
    if (i < n) xb[i] = f2bf(x[i]);
}

// ---------- 2) convert W -> W^T bf16 (wt[n*DD + k] = W[k*DD + n]) ----------
__global__ void k_cvt_wt(const float* __restrict__ W, unsigned short* __restrict__ wt) {
    int idx = blockIdx.x * 256 + threadIdx.x;    // 0 .. DD*DD-1, coalesced write
    int k = idx & (DD - 1);
    int n = idx >> 9;
    wt[idx] = f2bf(W[k * DD + n]);
}

// ---------- 3) GEMM1: h = X @ W  (bf16 WMMA, f32 acc) ----------
// also emits h^T in bf16 for GEMM2's B operand (K-contiguous per column)
__global__ void k_gemm1(const unsigned short* __restrict__ xb,
                        const unsigned short* __restrict__ wt,
                        float* __restrict__ h,
                        unsigned short* __restrict__ ht) {
    const int lane = threadIdx.x & 31;
    const int wave = threadIdx.x >> 5;
    const int m0   = blockIdx.x * 16;
    const int n0   = (blockIdx.y * 8 + wave) * 16;
    const int half = lane >> 4;          // 0|1
    const int l16  = lane & 15;

    const unsigned short* aRow = xb + (size_t)(m0 + l16) * DD + half * 8;   // A: K=half*8.., +16..
    const unsigned short* bRow = wt + (size_t)(n0 + l16) * DD + half * 8;   // B: K=half*8.., +16..

    v8f c = {};
#pragma unroll 4
    for (int k0 = 0; k0 < DD; k0 += 32) {
        ABFrag a, b;
        a.q[0] = *(const u32x4*)(aRow + k0);
        a.q[1] = *(const u32x4*)(aRow + k0 + 16);
        b.q[0] = *(const u32x4*)(bRow + k0);
        b.q[1] = *(const u32x4*)(bRow + k0 + 16);
        c = __builtin_amdgcn_wmma_f32_16x16x32_bf16(false, a.v, false, b.v,
                                                    (short)0, c, false, false);
    }
    // C layout: lane half*8 rows, column = n0+l16
    const int n     = n0 + l16;
    const int mbase = m0 + half * 8;
    Pack8 p8;
#pragma unroll
    for (int r = 0; r < 8; ++r) {
        float v = c[r];
        h[(size_t)(mbase + r) * DD + n] = v;
        p8.s[r] = f2bf(v);
    }
    *(u32x4*)(ht + (size_t)n * NN + mbase) = p8.q;   // 8 contiguous bf16 along M
}

// ---------- 4) e1 = h@a1, e2 = h@a2 (wave per row) ----------
__global__ void k_escore(const float* __restrict__ h, const float* __restrict__ a1,
                         const float* __restrict__ a2, float* __restrict__ e1,
                         float* __restrict__ e2) {
    const int lane = threadIdx.x & 31;
    const int wave = threadIdx.x >> 5;
    const int row  = blockIdx.x * 8 + wave;
    const float* hr = h + (size_t)row * DD;
    float s1 = 0.f, s2 = 0.f;
#pragma unroll
    for (int t = 0; t < DD; t += 32) {
        float v = hr[t + lane];
        s1 += v * a1[t + lane];
        s2 += v * a2[t + lane];
    }
#pragma unroll
    for (int off = 16; off > 0; off >>= 1) {
        s1 += __shfl_xor(s1, off, 32);
        s2 += __shfl_xor(s2, off, 32);
    }
    if (lane == 0) { e1[row] = s1; e2[row] = s2; }
}

// ---------- 5) per-row softmax stats: rowmax, 1/rowsum ----------
__global__ void k_stats(const float* __restrict__ adj, const float* __restrict__ e1,
                        const float* __restrict__ e2, float* __restrict__ rowmax,
                        float* __restrict__ rowinv) {
    __shared__ float red[256];
    const int row = blockIdx.x;
    const int tid = threadIdx.x;
    const float e1i = e1[row];
    const float* ar = adj + (size_t)row * NN;
    float sv[32];
    float mx = -3.0e38f;
#pragma unroll
    for (int t = 0; t < 32; ++t) {
        int j = tid + t * 256;
        float a = ar[j];
        float s = e1i + e2[j];
        s = (s >= 0.f) ? s : GALPHA * s;           // leaky_relu
        s = (a > -1.0e19f) ? s : NEGFILL;          // mask non-edges
        sv[t] = s;
        mx = fmaxf(mx, s);
    }
    red[tid] = mx; __syncthreads();
    for (int off = 128; off > 0; off >>= 1) {
        if (tid < off) red[tid] = fmaxf(red[tid], red[tid + off]);
        __syncthreads();
    }
    const float m = red[0];
    __syncthreads();
    float l = 0.f;
#pragma unroll
    for (int t = 0; t < 32; ++t) l += __expf(sv[t] - m);
    red[tid] = l; __syncthreads();
    for (int off = 128; off > 0; off >>= 1) {
        if (tid < off) red[tid] += red[tid + off];
        __syncthreads();
    }
    if (tid == 0) { rowmax[row] = m; rowinv[row] = 1.0f / red[0]; }
}

// ---------- 6) fused masked-softmax @ h, tanh epilogue ----------
// block = 16 output rows x 512 cols; 8 waves x 4 C-tiles.
// K step = 64: double-buffered 16x64 bf16 P tile in LDS, ONE barrier per step,
// 8 WMMAs per step per wave.
__global__ void k_gemm2(const float* __restrict__ adj, const float* __restrict__ e1,
                        const float* __restrict__ e2, const float* __restrict__ rowmax,
                        const float* __restrict__ rowinv,
                        const unsigned short* __restrict__ ht,
                        float* __restrict__ out) {
    __shared__ __align__(16) unsigned short Plds[2][16 * 64];
    const int tid  = threadIdx.x;
    const int lane = tid & 31;
    const int wave = tid >> 5;
    const int i0   = blockIdx.x * 16;

    // P-producer role: row pr, 4-wide column group pc
    const int pr = tid >> 4;            // 0..15
    const int pc = (tid & 15) * 4;      // 0..60
    const float e1i = e1[i0 + pr];
    const float mi  = rowmax[i0 + pr];
    const float li  = rowinv[i0 + pr];
    const float* arow = adj + (size_t)(i0 + pr) * NN;

    // WMMA consumer role
    const int half = lane >> 4;
    const int l16  = lane & 15;
    const unsigned short* bp0 = ht + (size_t)(wave * 64 +  0 + l16) * NN + half * 8;
    const unsigned short* bp1 = ht + (size_t)(wave * 64 + 16 + l16) * NN + half * 8;
    const unsigned short* bp2 = ht + (size_t)(wave * 64 + 32 + l16) * NN + half * 8;
    const unsigned short* bp3 = ht + (size_t)(wave * 64 + 48 + l16) * NN + half * 8;

    v8f c0 = {}, c1 = {}, c2 = {}, c3 = {};
    int it = 0;
    for (int k0 = 0; k0 < NN; k0 += 64, ++it) {
        unsigned short* buf = &Plds[it & 1][0];
        // --- build 16x64 P tile (bf16, normalized): 4 values per thread ---
        float4 av = *(const float4*)(arow + k0 + pc);
        float4 ev = *(const float4*)(e2 + k0 + pc);
        float s0 = e1i + ev.x; s0 = (s0 >= 0.f) ? s0 : GALPHA * s0;
        float s1 = e1i + ev.y; s1 = (s1 >= 0.f) ? s1 : GALPHA * s1;
        float s2 = e1i + ev.z; s2 = (s2 >= 0.f) ? s2 : GALPHA * s2;
        float s3 = e1i + ev.w; s3 = (s3 >= 0.f) ? s3 : GALPHA * s3;
        s0 = (av.x > -1.0e19f) ? s0 : NEGFILL;
        s1 = (av.y > -1.0e19f) ? s1 : NEGFILL;
        s2 = (av.z > -1.0e19f) ? s2 : NEGFILL;
        s3 = (av.w > -1.0e19f) ? s3 : NEGFILL;
        float p0 = __expf(s0 - mi) * li;
        float p1 = __expf(s1 - mi) * li;
        float p2 = __expf(s2 - mi) * li;
        float p3 = __expf(s3 - mi) * li;
        u32x2 pk;
        pk.x = ((unsigned int)f2bf(p1) << 16) | (unsigned int)f2bf(p0);
        pk.y = ((unsigned int)f2bf(p3) << 16) | (unsigned int)f2bf(p2);
        *(u32x2*)(buf + pr * 64 + pc) = pk;
        __syncthreads();   // single barrier: double-buffer makes it sufficient
        // --- consume: two A frags from LDS, 4 B tiles x 2 K-chunks from h^T ---
        const unsigned short* aL = buf + l16 * 64 + half * 8;
        ABFrag a0, a1, b0, b1, b2, b3;
        a0.q[0] = *(const u32x4*)(aL);      a0.q[1] = *(const u32x4*)(aL + 16);
        a1.q[0] = *(const u32x4*)(aL + 32); a1.q[1] = *(const u32x4*)(aL + 48);
        // K-chunk 0
        b0.q[0] = *(const u32x4*)(bp0 + k0); b0.q[1] = *(const u32x4*)(bp0 + k0 + 16);
        b1.q[0] = *(const u32x4*)(bp1 + k0); b1.q[1] = *(const u32x4*)(bp1 + k0 + 16);
        b2.q[0] = *(const u32x4*)(bp2 + k0); b2.q[1] = *(const u32x4*)(bp2 + k0 + 16);
        b3.q[0] = *(const u32x4*)(bp3 + k0); b3.q[1] = *(const u32x4*)(bp3 + k0 + 16);
        c0 = __builtin_amdgcn_wmma_f32_16x16x32_bf16(false, a0.v, false, b0.v, (short)0, c0, false, false);
        c1 = __builtin_amdgcn_wmma_f32_16x16x32_bf16(false, a0.v, false, b1.v, (short)0, c1, false, false);
        c2 = __builtin_amdgcn_wmma_f32_16x16x32_bf16(false, a0.v, false, b2.v, (short)0, c2, false, false);
        c3 = __builtin_amdgcn_wmma_f32_16x16x32_bf16(false, a0.v, false, b3.v, (short)0, c3, false, false);
        // K-chunk 1
        b0.q[0] = *(const u32x4*)(bp0 + k0 + 32); b0.q[1] = *(const u32x4*)(bp0 + k0 + 48);
        b1.q[0] = *(const u32x4*)(bp1 + k0 + 32); b1.q[1] = *(const u32x4*)(bp1 + k0 + 48);
        b2.q[0] = *(const u32x4*)(bp2 + k0 + 32); b2.q[1] = *(const u32x4*)(bp2 + k0 + 48);
        b3.q[0] = *(const u32x4*)(bp3 + k0 + 32); b3.q[1] = *(const u32x4*)(bp3 + k0 + 48);
        c0 = __builtin_amdgcn_wmma_f32_16x16x32_bf16(false, a1.v, false, b0.v, (short)0, c0, false, false);
        c1 = __builtin_amdgcn_wmma_f32_16x16x32_bf16(false, a1.v, false, b1.v, (short)0, c1, false, false);
        c2 = __builtin_amdgcn_wmma_f32_16x16x32_bf16(false, a1.v, false, b2.v, (short)0, c2, false, false);
        c3 = __builtin_amdgcn_wmma_f32_16x16x32_bf16(false, a1.v, false, b3.v, (short)0, c3, false, false);
    }
    // epilogue: tanh, strided f32 stores
    const int mb = i0 + half * 8;
#pragma unroll
    for (int r = 0; r < 8; ++r) {
        out[(size_t)(mb + r) * DD + (wave * 64 +  0 + l16)] = tanhf(c0[r]);
        out[(size_t)(mb + r) * DD + (wave * 64 + 16 + l16)] = tanhf(c1[r]);
        out[(size_t)(mb + r) * DD + (wave * 64 + 32 + l16)] = tanhf(c2[r]);
        out[(size_t)(mb + r) * DD + (wave * 64 + 48 + l16)] = tanhf(c3[r]);
    }
}

// ---------- launcher ----------
extern "C" void kernel_launch(void* const* d_in, const int* in_sizes, int n_in,
                              void* d_out, int out_size, void* d_ws, size_t ws_size,
                              hipStream_t stream) {
    (void)in_sizes; (void)n_in; (void)out_size; (void)ws_size;
    const float* x   = (const float*)d_in[0];   // [8192,512]
    const float* adj = (const float*)d_in[1];   // [8192,8192]
    const float* W   = (const float*)d_in[2];   // [512,512]
    const float* a1  = (const float*)d_in[3];   // [512]
    const float* a2  = (const float*)d_in[4];   // [512]
    float* out = (float*)d_out;

    char* ws = (char*)d_ws;
    size_t off = 0;
    auto carve = [&](size_t bytes) { void* p = ws + off; off += (bytes + 255) & ~(size_t)255; return p; };
    unsigned short* xb   = (unsigned short*)carve((size_t)NN * DD * 2);  // X bf16
    unsigned short* wt   = (unsigned short*)carve((size_t)DD * DD * 2);  // W^T bf16
    float*          h    = (float*)         carve((size_t)NN * DD * 4);  // h f32
    unsigned short* ht   = (unsigned short*)carve((size_t)DD * NN * 2);  // h^T bf16
    float*          e1   = (float*)         carve((size_t)NN * 4);
    float*          e2   = (float*)         carve((size_t)NN * 4);
    float*          rmax = (float*)         carve((size_t)NN * 4);
    float*          rinv = (float*)         carve((size_t)NN * 4);

    k_cvt_x  <<<(NN * DD) / 256, 256, 0, stream>>>(x, xb, NN * DD);
    k_cvt_wt <<<(DD * DD) / 256, 256, 0, stream>>>(W, wt);
    k_gemm1  <<<dim3(NN / 16, 4), 256, 0, stream>>>(xb, wt, h, ht);
    k_escore <<<NN / 8, 256, 0, stream>>>(h, a1, a2, e1, e2);
    k_stats  <<<NN, 256, 0, stream>>>(adj, e1, e2, rmax, rinv);
    k_gemm2  <<<NN / 16, 256, 0, stream>>>(adj, e1, e2, rmax, rinv, ht, out);
}
